// AttentionHead_69922067579438
// MI455X (gfx1250) — matrix operand order repaired and checked by reading, not compiled, not run on hardware
//
#include <hip/hip_runtime.h>

#define BB 32
#define TT 1024
#define CC 768
#define DD 64

typedef __attribute__((ext_vector_type(16))) _Float16 v16h;
typedef __attribute__((ext_vector_type(8)))  _Float16 v8h;
typedef __attribute__((ext_vector_type(4)))  _Float16 v4h;
typedef __attribute__((ext_vector_type(8)))  float    v8f;

static __device__ __forceinline__ v8f wmma_f16(v16h a, v16h b, v8f c) {
  // D = A(16x32 f16) * B(32x16 f16) + C(16x16 f32)
  return __builtin_amdgcn_wmma_f32_16x16x32_f16(false, a, false, b, (short)0, c,
                                                false, false);
}

// ---------------------------------------------------------------- kernel 0
__global__ void cvt_weights(const float* __restrict__ wq,
                            const float* __restrict__ wk,
                            const float* __restrict__ wv,
                            _Float16* __restrict__ wc) {
  int i = blockIdx.x * blockDim.x + threadIdx.x;
  if (i < DD * CC) {
    wc[i]               = (_Float16)wq[i];
    wc[DD * CC + i]     = (_Float16)wk[i];
    wc[2 * DD * CC + i] = (_Float16)wv[i];
  }
}

// ---------------------------------------------------------------- kernel 1
// One block = 16 rows of x (M-tile). 4 waves; wave w computes N-tiles
// {w, w+4, w+8} of the fused [Q|K|V] (192-dim) output. K = 768 = 24 steps.
__global__ void __launch_bounds__(128) qkv_proj(
    const float* __restrict__ x, const _Float16* __restrict__ wc,
    _Float16* __restrict__ qb, _Float16* __restrict__ kb,
    _Float16* __restrict__ vt) {
  __shared__ __align__(32) _Float16 xs[16 * CC];

  const int tid = threadIdx.x;
  const int m0  = blockIdx.x * 16;

  // Cooperative load of x tile, f32 -> f16, into LDS.
  const float4* xv = (const float4*)(x + (size_t)m0 * CC);
  for (int i = tid; i < 16 * CC / 4; i += 128) {
    __builtin_prefetch(&xv[i + 256], 0, 0);   // stream-ahead hint
    float4 v = xv[i];
    v4h h = {(_Float16)v.x, (_Float16)v.y, (_Float16)v.z, (_Float16)v.w};
    *(v4h*)(xs + i * 4) = h;
  }
  __syncthreads();

  const int lane = tid & 31;
  const int w    = tid >> 5;
  const int nl   = lane & 15;
  const int row  = lane & 15;                 // A-fragment: lane -> M row
  const int off0 = (lane & 16) ? 16 : 0;      // A: K-half split
  const int bo16 = (lane & 16) ? 16 : 0;      // B: K-half split
  const int hi8  = (lane & 16) ? 8 : 0;       // C: M offset for hi lanes

  v8f acc0 = {}, acc1 = {}, acc2 = {};

  for (int ks = 0; ks < CC; ks += 32) {
    // A fragment (16x32) from LDS: two contiguous 16B chunks per lane.
    v16h a;
    const _Float16* ap = xs + row * CC + ks + off0;
    ((v8h*)&a)[0] = *(const v8h*)(ap);
    ((v8h*)&a)[1] = *(const v8h*)(ap + 8);
    // B fragments (32x16) from Wc: per lane 16 contiguous K values.
    const _Float16* bbase = wc + (size_t)nl * CC + ks + bo16;
    v16h b0 = *(const v16h*)(bbase + (size_t)(w * 16) * CC);         // Q dims
    v16h b1 = *(const v16h*)(bbase + (size_t)((w + 4) * 16) * CC);   // K dims
    v16h b2 = *(const v16h*)(bbase + (size_t)((w + 8) * 16) * CC);   // V dims
    acc0 = wmma_f16(a, b0, acc0);
    acc1 = wmma_f16(a, b1, acc1);
    acc2 = wmma_f16(a, b2, acc2);
  }

  // Store Q (scaled by 1/sqrt(D)) and K row-major [m][d].
#pragma unroll
  for (int r = 0; r < 8; ++r) {
    const int m = m0 + r + hi8;
    qb[(size_t)m * DD + w * 16 + nl] = (_Float16)(acc0[r] * 0.125f);
    kb[(size_t)m * DD + w * 16 + nl] = (_Float16)(acc1[r]);
  }
  // Store V transposed: Vt[d][m] (contiguous 16B per lane).
  v8h vv;
#pragma unroll
  for (int r = 0; r < 8; ++r) vv[r] = (_Float16)acc2[r];
  *(v8h*)(vt + (size_t)(w * 16 + nl) * (BB * TT) + m0 + hi8) = vv;
}

// ---------------------------------------------------------------- kernel 2
// Flash attention. Block = (batch b, 64 query rows). 4 waves; wave w owns
// 16 query rows. Iterates 32-key tiles up to the causal diagonal.
__global__ void __launch_bounds__(128) flash_attn(
    const _Float16* __restrict__ qb, const _Float16* __restrict__ kb,
    const _Float16* __restrict__ vt, float* __restrict__ out) {
  __shared__ __align__(32) _Float16 ps[4][16 * 32];   // per-wave P scratch

  const int tid   = threadIdx.x;
  const int lane  = tid & 31;
  const int w     = tid >> 5;
  const int b     = blockIdx.y;
  const int qbase = blockIdx.x * 64;
  const int qw    = qbase + w * 16;
  const int nl    = lane & 15;
  const int off0  = (lane & 16) ? 16 : 0;
  const int bo16  = (lane & 16) ? 16 : 0;
  const int hi8   = (lane & 16) ? 8 : 0;

  // Q fragments for this wave's 16 rows (D=64 -> two 16x32 A-fragments).
  const _Float16* qp = qb + (size_t)(b * TT + qw + nl) * DD;
  v16h qf0, qf1;
  ((v8h*)&qf0)[0] = *(const v8h*)(qp + off0);
  ((v8h*)&qf0)[1] = *(const v8h*)(qp + off0 + 8);
  ((v8h*)&qf1)[0] = *(const v8h*)(qp + 32 + off0);
  ((v8h*)&qf1)[1] = *(const v8h*)(qp + 32 + off0 + 8);

  v8f o0 = {}, o1 = {}, o2 = {}, o3 = {};
  float mi[8], li[8];
#pragma unroll
  for (int r = 0; r < 8; ++r) { mi[r] = -1e30f; li[r] = 0.0f; }

  const float LOG2E = 1.4426950408889634f;
  const int nkt = (qbase + 64) / 32;
  _Float16* psw = ps[w];

  for (int kt = 0; kt < nkt; ++kt) {
    const int k0 = kt * 32;
    const bool active = (k0 <= qw + 15);   // wave-uniform causal skip
    if (active) {
      // ---- S = Q * K^T for keys [k0, k0+31] (two 16-key N-tiles)
      const _Float16* kp0 = kb + (size_t)(b * TT + k0 + nl) * DD + bo16;
      const _Float16* kp1 = kp0 + (size_t)16 * DD;
      v8f s0 = {}, s1 = {};
      s0 = wmma_f16(qf0, *(const v16h*)(kp0),      s0);
      s0 = wmma_f16(qf1, *(const v16h*)(kp0 + 32), s0);
      s1 = wmma_f16(qf0, *(const v16h*)(kp1),      s1);
      s1 = wmma_f16(qf1, *(const v16h*)(kp1 + 32), s1);

      // ---- causal mask + online softmax (f32)
      float p0[8], p1[8];
#pragma unroll
      for (int r = 0; r < 8; ++r) {
        const int qg = qw + r + hi8;
        if (k0 + nl      > qg) s0[r] = -1e30f;
        if (k0 + 16 + nl > qg) s1[r] = -1e30f;
        float tm = fmaxf(s0[r], s1[r]);
        tm = fmaxf(tm, __shfl_xor(tm, 1, 32));
        tm = fmaxf(tm, __shfl_xor(tm, 2, 32));
        tm = fmaxf(tm, __shfl_xor(tm, 4, 32));
        tm = fmaxf(tm, __shfl_xor(tm, 8, 32));
        const float mnew  = fmaxf(mi[r], tm);
        const float alpha = exp2f((mi[r] - mnew) * LOG2E);
        p0[r] = exp2f((s0[r] - mnew) * LOG2E);
        p1[r] = exp2f((s1[r] - mnew) * LOG2E);
        float rs = p0[r] + p1[r];
        rs += __shfl_xor(rs, 1, 32);
        rs += __shfl_xor(rs, 2, 32);
        rs += __shfl_xor(rs, 4, 32);
        rs += __shfl_xor(rs, 8, 32);
        li[r] = li[r] * alpha + rs;
        mi[r] = mnew;
        o0[r] *= alpha; o1[r] *= alpha; o2[r] *= alpha; o3[r] *= alpha;
      }

      // ---- C-layout -> A-layout transpose of P through per-wave LDS
      //      (DS ops are in-order per wave; no cross-wave sharing)
#pragma unroll
      for (int r = 0; r < 8; ++r) {
        const int qr = r + hi8;
        psw[qr * 32 + nl]      = (_Float16)p0[r];
        psw[qr * 32 + 16 + nl] = (_Float16)p1[r];
      }
      v16h pf;
      const _Float16* pp = psw + (lane & 15) * 32 + off0;
      ((v8h*)&pf)[0] = *(const v8h*)(pp);
      ((v8h*)&pf)[1] = *(const v8h*)(pp + 8);

      // ---- O += P * V  (Vt is dim-major -> contiguous B fragments)
      const _Float16* vb = vt + (size_t)nl * (BB * TT) + b * TT + k0 + bo16;
      const size_t vs = (size_t)16 * (BB * TT);
      o0 = wmma_f16(pf, *(const v16h*)(vb),          o0);
      o1 = wmma_f16(pf, *(const v16h*)(vb + vs),     o1);
      o2 = wmma_f16(pf, *(const v16h*)(vb + 2 * vs), o2);
      o3 = wmma_f16(pf, *(const v16h*)(vb + 3 * vs), o3);
    }
  }

  // ---- normalize and write O (f32)
  float* op = out + (size_t)b * TT * DD;
#pragma unroll
  for (int r = 0; r < 8; ++r) {
    const float inv = 1.0f / li[r];
    const int t = qw + r + hi8;
    op[(size_t)t * DD + nl]      = o0[r] * inv;
    op[(size_t)t * DD + 16 + nl] = o1[r] * inv;
    op[(size_t)t * DD + 32 + nl] = o2[r] * inv;
    op[(size_t)t * DD + 48 + nl] = o3[r] * inv;
  }
}

// ---------------------------------------------------------------- launch
extern "C" void kernel_launch(void* const* d_in, const int* in_sizes, int n_in,
                              void* d_out, int out_size, void* d_ws,
                              size_t ws_size, hipStream_t stream) {
  (void)in_sizes; (void)n_in; (void)out_size; (void)ws_size;
  const float* x  = (const float*)d_in[0];
  const float* wq = (const float*)d_in[1];
  const float* wk = (const float*)d_in[2];
  const float* wv = (const float*)d_in[3];
  float* out = (float*)d_out;

  // Workspace layout (f16): Wc[192*768] | Qb[M*64] | Kb[M*64] | Vt[64*M]
  _Float16* wc = (_Float16*)d_ws;
  _Float16* qbuf = wc + (size_t)3 * DD * CC;
  _Float16* kbuf = qbuf + (size_t)BB * TT * DD;
  _Float16* vbuf = kbuf + (size_t)BB * TT * DD;

  cvt_weights<<<(DD * CC + 255) / 256, 256, 0, stream>>>(wq, wk, wv, wc);
  qkv_proj<<<BB * TT / 16, 128, 0, stream>>>(x, wc, qbuf, kbuf, vbuf);
  flash_attn<<<dim3(TT / 64, BB), 128, 0, stream>>>(qbuf, kbuf, vbuf, out);
}
